// EGNNNet_8504035246170
// MI455X (gfx1250) — matrix operand order repaired and checked by reading, not compiled, not run on hardware
//
#include <hip/hip_runtime.h>
#include <hip/hip_bf16.h>

// CDNA5 / gfx1250 EGNN (2-layer edge-conditioned conv, scatter-max aggregation)
//
// Layer pipeline (bf16 WMMA, fp32 accumulate):
//   node GEMM :  M = A @ W_msg,  R = A @ W_root + b,  AGG <- key(-inf)
//                (layer 2 builds A on the fly: decode(AGG) + R_prev -> lrelu)
//   edge pass :  eproj = edge_attr @ W_edge   (v_wmma, K = DE = 32)
//                AGG[dst] = max(AGG[dst], key(eproj + M[src]))
//   finalize  :  out = lrelu(decode(AGG) + R)
//
// AGG holds order-preserving u32 keys (key = b ^ ((b>>31)|0x80000000), one
// v_bitop3), so scatter-max is a single branchless global_atomic_max_u32.
// edge_attr is read with NT hints so the 192MB L2 stays dedicated to M.

typedef __attribute__((ext_vector_type(16))) __bf16 v16bf;
typedef __attribute__((ext_vector_type(8)))  float  v8f;
typedef __attribute__((ext_vector_type(4)))  float  fv4;   // clang vec for NT loads

#define DOUTC 96   // output / node feature dim (6 N-tiles of 16)
#define KNODE 96   // node GEMM K               (3 K-tiles of 32)
#define DEC   32   // edge feature dim          (1 K-tile of 32)

#define EMPTY_KEY 0x007FFFFFu   // key(-inf): marks "no in-edges"

__device__ __forceinline__ unsigned fkey(float f) {
    int b = __float_as_int(f);
    return (unsigned)(b ^ ((b >> 31) | 0x80000000));
}
__device__ __forceinline__ float fdec(unsigned k) {
    return __int_as_float((k & 0x80000000u) ? (int)(k ^ 0x80000000u) : (int)~k);
}
__device__ __forceinline__ float aggval(unsigned k) {      // -inf (empty) -> 0
    return (k == EMPTY_KEY) ? 0.0f : fdec(k);
}
__device__ __forceinline__ float lrelu(float v) { return v > 0.0f ? v : 0.01f * v; }

// ---------------------------------------------------------------------------
// Node-level dual GEMM + bias + AGG-key init.  256 thr = 8 waves; each wave
// makes a 16-row x 96-col strip for BOTH W_msg and W_root (shared A frags).
// FUSED: A rows are produced from previous layer's (AGG keys, R) in-place.
// ---------------------------------------------------------------------------
template <bool FUSED>
__global__ __launch_bounds__(256) void egnn_node_gemm(
    const float*    __restrict__ Xin,   // !FUSED: node features [N,96]
    const unsigned*              AGGin, //  FUSED: prev agg keys [N,96] (== AGG)
    const float*                 Rin,   //  FUSED: prev root term [N,96] (== R)
    const float*    __restrict__ Wmsg,  // [96,96]
    const float*    __restrict__ Wroot, // [96,96]
    const float*    __restrict__ bias,  // [96]
    float*          __restrict__ M,     // [N,96]
    float*                       R,     // [N,96]
    unsigned*                    AGG,   // [N,96] <- EMPTY_KEY
    int N)
{
    // Weights staged in WMMA-B fragment order [kt][nt][lane][i]:
    // one contiguous 32B LDS read per fragment per lane.
    __shared__ __align__(32) __bf16 sWm[3 * 6 * 32 * 16];
    __shared__ __align__(32) __bf16 sWr[3 * 6 * 32 * 16];
    __shared__ float sBias[DOUTC];

    for (int idx = threadIdx.x; idx < 3 * 6 * 32 * 16; idx += 256) {
        int i    = idx & 15;
        int lane = (idx >> 4) & 31;
        int t    = idx >> 9;                         // kt*6 + nt
        int nt   = t % 6, kt = t / 6;
        int k = kt * 32 + ((i >> 3) << 4) + ((lane >> 4) << 3) + (i & 7);    // B: K = (i>>3)*16 + (lane>>4)*8 + (i&7)
        int n = nt * 16 + (lane & 15);               // B: col = lane&15
        sWm[idx] = (__bf16)Wmsg[k * DOUTC + n];
        sWr[idx] = (__bf16)Wroot[k * DOUTC + n];
    }
    if (threadIdx.x < DOUTC) sBias[threadIdx.x] = bias[threadIdx.x];
    __syncthreads();

    const int wave  = threadIdx.x >> 5;
    const int lane  = threadIdx.x & 31;
    const int rbase = blockIdx.x * 128 + wave * 16;
    if (rbase >= N) return;

    const int  arow  = rbase + (lane & 15);
    const int  koffL = (lane >> 4) * 8;
    const int  colL  = lane & 15;
    const int  rowHf = (lane >> 4) * 8;
    const bool full  = (rbase + 16 <= N);
    const bool ain   = (arow < N);

    float bv[6];
    #pragma unroll
    for (int nt = 0; nt < 6; ++nt) bv[nt] = sBias[nt * 16 + colL];

    // ---- A fragments: 3 K-tiles of the 16-row strip (ISA bf16 16x32 layout)
    v16bf a[3];
    #pragma unroll
    for (int kt = 0; kt < 3; ++kt) {
        if (ain) {
            const unsigned roff = (unsigned)arow * KNODE + kt * 32 + koffL;
            if (FUSED) {
                const uint4 k0 = *(const uint4*)(AGGin + roff);
                const uint4 k1 = *(const uint4*)(AGGin + roff + 4);
                const uint4 k2 = *(const uint4*)(AGGin + roff + 16);
                const uint4 k3 = *(const uint4*)(AGGin + roff + 20);
                const fv4   r0 = *(const fv4*)(Rin + roff);
                const fv4   r1 = *(const fv4*)(Rin + roff + 4);
                const fv4   r2 = *(const fv4*)(Rin + roff + 16);
                const fv4   r3 = *(const fv4*)(Rin + roff + 20);
                a[kt][0]  = (__bf16)lrelu(aggval(k0.x) + r0.x);
                a[kt][1]  = (__bf16)lrelu(aggval(k0.y) + r0.y);
                a[kt][2]  = (__bf16)lrelu(aggval(k0.z) + r0.z);
                a[kt][3]  = (__bf16)lrelu(aggval(k0.w) + r0.w);
                a[kt][4]  = (__bf16)lrelu(aggval(k1.x) + r1.x);
                a[kt][5]  = (__bf16)lrelu(aggval(k1.y) + r1.y);
                a[kt][6]  = (__bf16)lrelu(aggval(k1.z) + r1.z);
                a[kt][7]  = (__bf16)lrelu(aggval(k1.w) + r1.w);
                a[kt][8]  = (__bf16)lrelu(aggval(k2.x) + r2.x);
                a[kt][9]  = (__bf16)lrelu(aggval(k2.y) + r2.y);
                a[kt][10] = (__bf16)lrelu(aggval(k2.z) + r2.z);
                a[kt][11] = (__bf16)lrelu(aggval(k2.w) + r2.w);
                a[kt][12] = (__bf16)lrelu(aggval(k3.x) + r3.x);
                a[kt][13] = (__bf16)lrelu(aggval(k3.y) + r3.y);
                a[kt][14] = (__bf16)lrelu(aggval(k3.z) + r3.z);
                a[kt][15] = (__bf16)lrelu(aggval(k3.w) + r3.w);
            } else {
                const fv4 l0 = *(const fv4*)(Xin + roff);
                const fv4 l1 = *(const fv4*)(Xin + roff + 4);
                const fv4 h0 = *(const fv4*)(Xin + roff + 16);
                const fv4 h1 = *(const fv4*)(Xin + roff + 20);
                a[kt][0]=(__bf16)l0.x; a[kt][1]=(__bf16)l0.y; a[kt][2]=(__bf16)l0.z; a[kt][3]=(__bf16)l0.w;
                a[kt][4]=(__bf16)l1.x; a[kt][5]=(__bf16)l1.y; a[kt][6]=(__bf16)l1.z; a[kt][7]=(__bf16)l1.w;
                a[kt][8]=(__bf16)h0.x; a[kt][9]=(__bf16)h0.y; a[kt][10]=(__bf16)h0.z; a[kt][11]=(__bf16)h0.w;
                a[kt][12]=(__bf16)h1.x; a[kt][13]=(__bf16)h1.y; a[kt][14]=(__bf16)h1.z; a[kt][15]=(__bf16)h1.w;
            }
        } else {
            #pragma unroll
            for (int i = 0; i < 16; ++i) a[kt][i] = (__bf16)0.0f;
        }
    }

    // per-lane base element offset; i*96 + nt*16 folds into ioffset.
    const unsigned base = (unsigned)(rbase + rowHf) * DOUTC + colL;

    #pragma unroll
    for (int nt = 0; nt < 6; ++nt) {
        v8f cm = {}, cr = {};
        #pragma unroll
        for (int kt = 0; kt < 3; ++kt) {
            v16bf bm = *(const v16bf*)&sWm[((kt * 6 + nt) * 32 + lane) * 16];
            v16bf br = *(const v16bf*)&sWr[((kt * 6 + nt) * 32 + lane) * 16];
            cm = __builtin_amdgcn_wmma_f32_16x16x32_bf16(false, a[kt], false, bm,
                                                         (short)0, cm, false, false);
            cr = __builtin_amdgcn_wmma_f32_16x16x32_bf16(false, a[kt], false, br,
                                                         (short)0, cr, false, false);
        }
        // C/D layout: reg i, lanes 0-15 -> row i; lanes 16-31 -> row 8+i
        if (full) {
            #pragma unroll
            for (int i = 0; i < 8; ++i) {
                const unsigned off = base + i * DOUTC + nt * 16;
                M[off]   = cm[i];
                R[off]   = cr[i] + bv[nt];
                AGG[off] = EMPTY_KEY;
            }
        } else {
            #pragma unroll
            for (int i = 0; i < 8; ++i) {
                if (rbase + rowHf + i < N) {
                    const unsigned off = base + i * DOUTC + nt * 16;
                    M[off]   = cm[i];
                    R[off]   = cr[i] + bv[nt];
                    AGG[off] = EMPTY_KEY;
                }
            }
        }
    }
}

// ---------------------------------------------------------------------------
// Fused edge projection (WMMA, K=32) + gather M[src] + branchless scatter-max.
// Each wave processes 16-edge tiles, grid-stride over E/16 tiles.
// ---------------------------------------------------------------------------
__global__ __launch_bounds__(256) void egnn_edge_scatter(
    const float* __restrict__ EA,     // [E,32]
    const int*   __restrict__ src,    // [E]
    const int*   __restrict__ dst,    // [E]
    const float* __restrict__ Wedge,  // [32,96]
    const float* __restrict__ M,      // [N,96]
    unsigned*                 AGG,    // [N,96] keys
    int E)
{
    __shared__ __align__(32) __bf16 sW[6 * 32 * 16];   // [nt][lane][i]
    for (int idx = threadIdx.x; idx < 6 * 32 * 16; idx += 256) {
        int i    = idx & 15;
        int lane = (idx >> 4) & 31;
        int nt   = idx >> 9;
        int k = ((i >> 3) << 4) + ((lane >> 4) << 3) + (i & 7);   // 0..31
        int n = nt * 16 + (lane & 15);
        sW[idx] = (__bf16)Wedge[k * DOUTC + n];
    }
    __syncthreads();

    const int wave   = threadIdx.x >> 5;
    const int lane   = threadIdx.x & 31;
    const int nTiles = (E + 15) >> 4;
    const int wId    = blockIdx.x * 8 + wave;
    const int nW     = gridDim.x * 8;

    // W_edge B fragments live in registers for the whole kernel
    v16bf b[6];
    #pragma unroll
    for (int nt = 0; nt < 6; ++nt)
        b[nt] = *(const v16bf*)&sW[(nt * 32 + lane) * 16];

    const int koffL = (lane >> 4) * 8;
    const int colL  = lane & 15;
    const int rowHf = (lane >> 4) * 8;

    for (int tile = wId; tile < nTiles; tile += nW) {
        const int  ebase = tile << 4;
        const bool fullT = (ebase + 16 <= E);

        if (tile + nW < nTiles)
            __builtin_prefetch(EA + ((unsigned)(ebase + (nW << 4)) + (lane & 15)) * DEC, 0, 0);

        // Hoist the 16 src/dst index loads so their latency hides under the
        // EA loads + WMMAs below.
        unsigned moff[8], aoff[8];
        if (fullT) {
            #pragma unroll
            for (int i = 0; i < 8; ++i) {
                const int e = ebase + rowHf + i;
                moff[i] = (unsigned)src[e] * DOUTC + colL;
                aoff[i] = (unsigned)dst[e] * DOUTC + colL;
            }
        }

        // A fragment = 16 edge_attr rows (K = DE = 32), streamed with NT hint
        v16bf a;
        const int er = ebase + (lane & 15);
        if (fullT || er < E) {
            const float* p = EA + (unsigned)er * DEC + koffL;
            const fv4 l0 = __builtin_nontemporal_load((const fv4*)(p));
            const fv4 l1 = __builtin_nontemporal_load((const fv4*)(p + 4));
            const fv4 h0 = __builtin_nontemporal_load((const fv4*)(p + 16));
            const fv4 h1 = __builtin_nontemporal_load((const fv4*)(p + 20));
            a[0]=(__bf16)l0.x; a[1]=(__bf16)l0.y; a[2]=(__bf16)l0.z; a[3]=(__bf16)l0.w;
            a[4]=(__bf16)l1.x; a[5]=(__bf16)l1.y; a[6]=(__bf16)l1.z; a[7]=(__bf16)l1.w;
            a[8]=(__bf16)h0.x; a[9]=(__bf16)h0.y; a[10]=(__bf16)h0.z; a[11]=(__bf16)h0.w;
            a[12]=(__bf16)h1.x; a[13]=(__bf16)h1.y; a[14]=(__bf16)h1.z; a[15]=(__bf16)h1.w;
        } else {
            #pragma unroll
            for (int i = 0; i < 16; ++i) a[i] = (__bf16)0.0f;
        }

        v8f acc[6];
        #pragma unroll
        for (int nt = 0; nt < 6; ++nt) {
            v8f c = {};
            acc[nt] = __builtin_amdgcn_wmma_f32_16x16x32_bf16(false, a, false, b[nt],
                                                              (short)0, c, false, false);
        }

        // Reg i of lane l = edge (rowHf+i), column (nt*16 + colL): a half-wave
        // reads 16 consecutive floats of one M row (coalesced 64B gather).
        if (fullT) {
            #pragma unroll
            for (int i = 0; i < 8; ++i) {
                float mv[6];
                #pragma unroll
                for (int nt = 0; nt < 6; ++nt)          // 6-load clause, 1 wait
                    mv[nt] = M[moff[i] + nt * 16];
                #pragma unroll
                for (int nt = 0; nt < 6; ++nt)          // 6 waitless atomics
                    atomicMax(AGG + aoff[i] + nt * 16, fkey(acc[nt][i] + mv[nt]));
            }
        } else {
            #pragma unroll
            for (int i = 0; i < 8; ++i) {
                const int e = ebase + rowHf + i;
                if (e < E) {
                    const unsigned mo = (unsigned)src[e] * DOUTC + colL;
                    const unsigned ao = (unsigned)dst[e] * DOUTC + colL;
                    float mv[6];
                    #pragma unroll
                    for (int nt = 0; nt < 6; ++nt) mv[nt] = M[mo + nt * 16];
                    #pragma unroll
                    for (int nt = 0; nt < 6; ++nt)
                        atomicMax(AGG + ao + nt * 16, fkey(acc[nt][i] + mv[nt]));
                }
            }
        }
    }
}

// ---------------------------------------------------------------------------
// Final: out = lrelu(decode(AGG) + R)   (single-use streams -> NT loads)
// ---------------------------------------------------------------------------
__global__ __launch_bounds__(256) void egnn_finalize(
    const unsigned* __restrict__ AGG,
    const float*    __restrict__ R,
    float*          __restrict__ out,
    long total)
{
    long i = (long)blockIdx.x * 256 + threadIdx.x;
    if (i < total) {
        const unsigned k = __builtin_nontemporal_load(AGG + i);
        const float    r = __builtin_nontemporal_load(R + i);
        out[i] = lrelu(aggval(k) + r);
    }
}

// ---------------------------------------------------------------------------
extern "C" void kernel_launch(void* const* d_in, const int* in_sizes, int n_in,
                              void* d_out, int out_size, void* d_ws, size_t ws_size,
                              hipStream_t stream) {
    const float* X   = (const float*)d_in[0];
    const int*   ei  = (const int*)  d_in[1];   // [2,E] row-major: src then dst
    const float* EA  = (const float*)d_in[2];
    const float* W1m = (const float*)d_in[3];
    const float* W1e = (const float*)d_in[4];
    const float* W1r = (const float*)d_in[5];
    const float* b1  = (const float*)d_in[6];
    const float* W2m = (const float*)d_in[7];
    const float* W2e = (const float*)d_in[8];
    const float* W2r = (const float*)d_in[9];
    const float* b2  = (const float*)d_in[10];

    const int  N   = in_sizes[0] / DOUTC;
    const int  E   = in_sizes[1] / 2;
    const int* src = ei;
    const int* dst = ei + E;

    float*    ws  = (float*)d_ws;
    const long nf = (long)N * DOUTC;
    float*    M   = ws;
    float*    R   = ws + nf;
    unsigned* AGG = (unsigned*)(ws + 2 * nf);

    const dim3 blk(256);
    const int gNode  = (N + 127) / 128;
    const int nTiles = (E + 15) >> 4;
    int gEdge = (nTiles + 7) / 8;
    if (gEdge > 1200) gEdge = 1200;
    if (gEdge < 1)    gEdge = 1;
    const int gElem = (int)((nf + 255) / 256);

    // Layer 1
    egnn_node_gemm<false><<<gNode, blk, 0, stream>>>(X, nullptr, nullptr,
                                                     W1m, W1r, b1, M, R, AGG, N);
    egnn_edge_scatter<<<gEdge, blk, 0, stream>>>(EA, src, dst, W1e, M, AGG, E);
    // Layer 2 (activation of layer 1 fused into the A-fragment load)
    egnn_node_gemm<true><<<gNode, blk, 0, stream>>>(nullptr, AGG, R,
                                                    W2m, W2r, b2, M, R, AGG, N);
    egnn_edge_scatter<<<gEdge, blk, 0, stream>>>(EA, src, dst, W2e, M, AGG, E);
    egnn_finalize<<<gElem, blk, 0, stream>>>(AGG, R, (float*)d_out, nf);
}
